// GroupedQueryAttention_76441827934739
// MI455X (gfx1250) — matrix operand order repaired and checked by reading, not compiled, not run on hardware
//
#include <hip/hip_runtime.h>

// ---------------------------------------------------------------------------
// GQA forward for MI455X (gfx1250, wave32, WMMA, double-buffered async
// global->LDS staging).
// B=2, L=2048, D=1024, H=16 query heads, KV=4 kv heads, Hd=64, group=4.
// Pipeline:
//   0) f32 -> bf16 conversion of x and all weights (once).
//   1) unified bf16 GEMM (double-buffered async-LDS tiles) for Q/K/V, O-proj.
//   2) flash-style attention, K/V tiles double-buffered async-staged in LDS.
// All matmuls via v_wmma_f32_16x16x32_bf16 (f32 accumulate).
// ---------------------------------------------------------------------------

typedef __attribute__((ext_vector_type(16))) __bf16          v16bf;
typedef __attribute__((ext_vector_type(8)))  float           v8f;
typedef __attribute__((ext_vector_type(4)))  float           v4f;
typedef __attribute__((ext_vector_type(8)))  unsigned short  ushortx8;
typedef __attribute__((ext_vector_type(4)))  unsigned short  ushortx4;

union Frag16 {
    v16bf v;
    unsigned short u[16];
};

static __device__ inline unsigned short f2bf(float f) {
    union { float f; unsigned u; } c; c.f = f;
    unsigned u = c.u;
    unsigned r = (u + 0x7fffu + ((u >> 16) & 1u)) >> 16;   // RNE
    return (unsigned short)r;
}

static __device__ inline v8f wmma_bf16(const Frag16& a, const Frag16& b, v8f c) {
    return __builtin_amdgcn_wmma_f32_16x16x32_bf16(false, a.v, false, b.v,
                                                   (short)0, c, false, false);
}

static __device__ inline v8f vzero8() {
    v8f z;
#pragma unroll
    for (int e = 0; e < 8; ++e) z[e] = 0.0f;
    return z;
}

// Async 16B global->LDS copy (ASYNCcnt-tracked). lds_off = byte offset in LDS.
static __device__ inline void async_copy_b128(unsigned lds_off, const void* gptr) {
    asm volatile("global_load_async_to_lds_b128 %0, %1, off"
                 :: "v"(lds_off), "v"((unsigned long long)(uintptr_t)gptr)
                 : "memory");
}
// Wait until ASYNCcnt <= N. Async loads complete in order per wave, so a
// partial wait releases exactly the oldest (current-tile) copies.
template <int N>
static __device__ inline void wait_asynccnt() {
    asm volatile("s_wait_asynccnt %0" :: "i"(N) : "memory");
}
static __device__ inline unsigned lds_addr(const void* p) {
    return (unsigned)(uintptr_t)p;      // AS(3) pointer -> LDS byte offset
}

// ---------------------------------------------------------------------------
// Kernel 0: f32 -> bf16 conversion (one float4 per thread).
// ---------------------------------------------------------------------------
__global__ void __launch_bounds__(256)
gqa_cvt_kernel(const float* __restrict__ src, unsigned short* __restrict__ dst,
               int n4) {
    int i = blockIdx.x * 256 + threadIdx.x;
    if (i < n4) {
        v4f f = ((const v4f*)src)[i];
        ushortx4 o;
#pragma unroll
        for (int e = 0; e < 4; ++e) o[e] = f2bf(f[e]);
        ((ushortx4*)dst)[i] = o;
    }
}

// ---------------------------------------------------------------------------
// Kernel 1: unified bf16 GEMM, out = A (4096 x 1024) * Bw^T (N x 1024).
// Block = 256 thr (8 waves, 4M x 2N), block tile 128x128, wave tile 32x64.
// 128x64 tiles of A and Bw double-buffered via async global->LDS; tile t+1
// streams in while tile t computes (16 WMMA per wave between barriers).
// mode 0: bf16 out, head-major [B, NH, L, 64]         (Q, K)
// mode 1: bf16 out, head-transposed [B, NH, 64, L]    (V)
// mode 2: f32 out, row-major [4096, N]                (final projection)
// ---------------------------------------------------------------------------

// Stage a 128-row x 64-col bf16 tile (global row stride 1024) into LDS.
// 1024 chunks of 16B; 4 chunks per thread.
static __device__ inline void stage_tile_128x64(unsigned lbase,
                                                const unsigned short* g,
                                                int k0, int tid) {
#pragma unroll
    for (int c = 0; c < 4; ++c) {
        const int idx = c * 256 + tid;          // 0..1023
        const int row = idx >> 3;
        const int col = (idx & 7) * 8;
        async_copy_b128(lbase + idx * 16, g + (size_t)row * 1024 + k0 + col);
    }
}

__global__ void __launch_bounds__(256)
gqa_gemm_kernel(const unsigned short* __restrict__ A,
                const unsigned short* __restrict__ Bw,
                void* __restrict__ out, int N, int NH, int mode) {
    const int lane  = threadIdx.x & 31;
    const int wid   = threadIdx.x >> 5;
    const int wm    = wid >> 1;
    const int wn    = wid & 1;
    const int lmod  = lane & 15;
    const int lhalf = lane >> 4;
    const int mRow0 = blockIdx.y * 128;
    const int nCol0 = blockIdx.x * 128;
    const unsigned short* Ablk = A  + (size_t)mRow0 * 1024;
    const unsigned short* Bblk = Bw + (size_t)nCol0 * 1024;

    __shared__ __align__(16) unsigned short abuf[2][128 * 64];   // 2 x 16 KB
    __shared__ __align__(16) unsigned short bbuf[2][128 * 64];   // 2 x 16 KB

    v8f acc[2][4];
#pragma unroll
    for (int i = 0; i < 2; ++i)
#pragma unroll
        for (int j = 0; j < 4; ++j) acc[i][j] = vzero8();

    // Prologue: stage tile 0 into buffer 0 (8 async ops / thread / tile).
    stage_tile_128x64(lds_addr(abuf[0]), Ablk, 0, threadIdx.x);
    stage_tile_128x64(lds_addr(bbuf[0]), Bblk, 0, threadIdx.x);

    const int T = 1024 / 64;     // 16 K-tiles
    for (int t = 0; t < T; ++t) {
        const int cur = t & 1;
        if (t + 1 < T) {   // prefetch next tile into alternate buffer
            stage_tile_128x64(lds_addr(abuf[cur ^ 1]), Ablk, (t + 1) * 64,
                              threadIdx.x);
            stage_tile_128x64(lds_addr(bbuf[cur ^ 1]), Bblk, (t + 1) * 64,
                              threadIdx.x);
            wait_asynccnt<8>();   // oldest 8 (= tile t) have landed
        } else {
            wait_asynccnt<0>();
        }
        __syncthreads();          // tile t visible to all waves

        const unsigned short* ab = abuf[cur];
        const unsigned short* bb = bbuf[cur];
#pragma unroll
        for (int ks = 0; ks < 2; ++ks) {       // two K=32 sub-steps
            Frag16 a[2];
#pragma unroll
            for (int i = 0; i < 2; ++i) {
                const unsigned short* r =
                    ab + (wm * 32 + i * 16 + lmod) * 64 + ks * 32;
                ushortx8 c0 = *(const ushortx8*)(r + lhalf * 8);
                ushortx8 c1 = *(const ushortx8*)(r + 16 + lhalf * 8);
#pragma unroll
                for (int e = 0; e < 8; ++e) { a[i].u[e] = c0[e]; a[i].u[8 + e] = c1[e]; }
            }
            Frag16 bfr[4];
#pragma unroll
            for (int j = 0; j < 4; ++j) {
                const unsigned short* r =
                    bb + (wn * 64 + j * 16 + lmod) * 64 + ks * 32;
                ushortx8 c0 = *(const ushortx8*)(r + lhalf * 8);
                ushortx8 c1 = *(const ushortx8*)(r + 16 + lhalf * 8);
#pragma unroll
                for (int e = 0; e < 8; ++e) { bfr[j].u[e] = c0[e]; bfr[j].u[8 + e] = c1[e]; }
            }
#pragma unroll
            for (int i = 0; i < 2; ++i)
#pragma unroll
                for (int j = 0; j < 4; ++j)
                    acc[i][j] = wmma_bf16(a[i], bfr[j], acc[i][j]);
        }
        __syncthreads();          // reads done before buffer cur is re-staged
    }

    const int mBase = mRow0 + wm * 32;
    const int nBase = nCol0 + wn * 64;
    if (mode == 2) {
        float* o = (float*)out;
#pragma unroll
        for (int i = 0; i < 2; ++i)
#pragma unroll
            for (int j = 0; j < 4; ++j) {
                const int n = nBase + j * 16 + lmod;
#pragma unroll
                for (int r = 0; r < 8; ++r) {
                    const int m = mBase + i * 16 + lhalf * 8 + r;
                    o[(size_t)m * N + n] = acc[i][j][r];
                }
            }
    } else if (mode == 1) {   // bf16, [B, NH, 64, L] (transposed V)
        unsigned short* o = (unsigned short*)out;
#pragma unroll
        for (int i = 0; i < 2; ++i)
#pragma unroll
            for (int j = 0; j < 4; ++j) {
                const int n0   = nBase + j * 16 + lmod;
                const int head = n0 >> 6;
                const int h    = n0 & 63;
#pragma unroll
                for (int r = 0; r < 8; ++r) {
                    const int mm = mBase + i * 16 + lhalf * 8 + r;
                    const int bb2 = mm >> 11;
                    const int lq  = mm & 2047;
                    o[(((size_t)bb2 * NH + head) * 64 + h) * 2048 + lq] =
                        f2bf(acc[i][j][r]);
                }
            }
    } else {                  // bf16, [B, NH, L, 64]
        unsigned short* o = (unsigned short*)out;
#pragma unroll
        for (int i = 0; i < 2; ++i)
#pragma unroll
            for (int j = 0; j < 4; ++j) {
                const int n0   = nBase + j * 16 + lmod;
                const int head = n0 >> 6;
                const int h    = n0 & 63;
#pragma unroll
                for (int r = 0; r < 8; ++r) {
                    const int mm = mBase + i * 16 + lhalf * 8 + r;
                    const int bb2 = mm >> 11;
                    const int lq  = mm & 2047;
                    o[(((size_t)bb2 * NH + head) * 2048 + lq) * 64 + h] =
                        f2bf(acc[i][j][r]);
                }
            }
    }
}

// ---------------------------------------------------------------------------
// Kernel 2: flash-style GQA attention over bf16 Q/K/V.
// Block = (b, head, 128-query chunk); 8 waves x 16 query rows each.
// K tile [key][h] and V^T tile [h][key] double-buffered async-staged in LDS
// (4 async ops / thread / key-block). P transposed through per-wave LDS.
// ---------------------------------------------------------------------------
__global__ void __launch_bounds__(256)
gqa_attn_kernel(const unsigned short* __restrict__ Q,
                const unsigned short* __restrict__ Kd,
                const unsigned short* __restrict__ Vt,
                const int* __restrict__ mask,
                unsigned short* __restrict__ O) {
    const int b    = blockIdx.z;
    const int head = blockIdx.y;
    const int kv   = head >> 2;          // group = 4
    const int wid  = threadIdx.x >> 5;
    const int lane = threadIdx.x & 31;
    const int lmod  = lane & 15;
    const int lhalf = lane >> 4;
    const int qBase = blockIdx.x * 128 + wid * 16;

    const unsigned short* Qh = Q  + ((size_t)(b * 16 + head) * 2048) * 64;
    const unsigned short* Kh = Kd + ((size_t)(b * 4  + kv)   * 2048) * 64;
    const unsigned short* Vh = Vt + ((size_t)(b * 4  + kv)   * 64)   * 2048;

    __shared__ __align__(16) unsigned short kbuf[2][64 * 64];  // 2 x 8 KB
    __shared__ __align__(16) unsigned short vbuf[2][64 * 64];  // 2 x 8 KB
    __shared__ __align__(16) unsigned short pts[8][16 * 64];   // 16 KB

    // Stage K tile ([key][h]) + V^T tile ([h][key]) for key block kb.
    auto stageKV = [&](int buf, int kb) {
        const unsigned kb_l = lds_addr(kbuf[buf]);
        const unsigned vb_l = lds_addr(vbuf[buf]);
#pragma unroll
        for (int c = 0; c < 2; ++c) {
            const int idx  = c * 256 + threadIdx.x;    // 0..511
            const int row  = idx >> 3;                 // 64 rows
            const int coff = (idx & 7) * 8;            // 8 chunks / 64-el row
            async_copy_b128(kb_l + idx * 16, Kh + (size_t)(kb + row) * 64 + coff);
            async_copy_b128(vb_l + idx * 16, Vh + (size_t)row * 2048 + kb + coff);
        }
    };

    // Q A-fragments (Hd=64 -> 2 fragments of 16x32), loaded once from global.
    Frag16 aq[2];
#pragma unroll
    for (int kt = 0; kt < 2; ++kt) {
        const unsigned short* row = Qh + (size_t)(qBase + lmod) * 64 + kt * 32;
        ushortx8 c0 = *(const ushortx8*)(row + lhalf * 8);
        ushortx8 c1 = *(const ushortx8*)(row + 16 + lhalf * 8);
#pragma unroll
        for (int e = 0; e < 8; ++e) { aq[kt].u[e] = c0[e]; aq[kt].u[8 + e] = c1[e]; }
    }

    v8f oacc[4];
#pragma unroll
    for (int j = 0; j < 4; ++j) oacc[j] = vzero8();
    float m_run[8], l_run[8];
#pragma unroll
    for (int r = 0; r < 8; ++r) { m_run[r] = -1e30f; l_run[r] = 0.0f; }

    stageKV(0, 0);                       // prologue
    const int T = 2048 / 64;             // 32 key blocks
    for (int t = 0; t < T; ++t) {
        const int cur = t & 1;
        const int kb  = t * 64;
        if (t + 1 < T) {
            stageKV(cur ^ 1, kb + 64);   // prefetch next block
            wait_asynccnt<4>();          // oldest 4 (= block t) have landed
        } else {
            wait_asynccnt<0>();
        }
        __syncthreads();

        const unsigned short* kbp = kbuf[cur];
        const unsigned short* vbp = vbuf[cur];

        // ---- S = Q K^T for a 16x64 score tile ----
        v8f s[4];
#pragma unroll
        for (int nt = 0; nt < 4; ++nt) s[nt] = vzero8();
#pragma unroll
        for (int nt = 0; nt < 4; ++nt) {
#pragma unroll
            for (int kt = 0; kt < 2; ++kt) {
                Frag16 bk;  // B[k=h, n=lk] = K[lk, h]
                const unsigned short* row =
                    kbp + (nt * 16 + lmod) * 64 + kt * 32;
                ushortx8 c0 = *(const ushortx8*)(row + lhalf * 8);
                ushortx8 c1 = *(const ushortx8*)(row + 16 + lhalf * 8);
#pragma unroll
                for (int e = 0; e < 8; ++e) { bk.u[e] = c0[e]; bk.u[8 + e] = c1[e]; }
                s[nt] = wmma_bf16(aq[kt], bk, s[nt]);
            }
        }

        // ---- scale 1/sqrt(64), mask, row-max ----
        float fm[8];
#pragma unroll
        for (int r = 0; r < 8; ++r) fm[r] = -1e30f;
#pragma unroll
        for (int nt = 0; nt < 4; ++nt) {
            const int key = kb + nt * 16 + lmod;
            const float bias = (mask[b * 2048 + key] != 0) ? 0.0f : -1e30f;
#pragma unroll
            for (int r = 0; r < 8; ++r) {
                float v = s[nt][r] * 0.125f + bias;
                s[nt][r] = v;
                fm[r] = fmaxf(fm[r], v);
            }
        }
#pragma unroll
        for (int off = 8; off >= 1; off >>= 1)
#pragma unroll
            for (int r = 0; r < 8; ++r)
                fm[r] = fmaxf(fm[r], __shfl_xor(fm[r], off, 32));

        float alpha[8], ls[8];
#pragma unroll
        for (int r = 0; r < 8; ++r) {
            float mn = fmaxf(m_run[r], fm[r]);
            alpha[r] = __expf(m_run[r] - mn);
            m_run[r] = mn;
            ls[r] = 0.0f;
        }

        // ---- P = exp(S - m): bf16 to per-wave LDS (C-layout -> A-layout) ----
        unsigned short* pw = pts[wid];
#pragma unroll
        for (int nt = 0; nt < 4; ++nt) {
#pragma unroll
            for (int r = 0; r < 8; ++r) {
                float p = __expf(s[nt][r] - m_run[r]);
                ls[r] += p;
                pw[(lhalf * 8 + r) * 64 + nt * 16 + lmod] = f2bf(p);
            }
        }
#pragma unroll
        for (int off = 8; off >= 1; off >>= 1)
#pragma unroll
            for (int r = 0; r < 8; ++r)
                ls[r] += __shfl_xor(ls[r], off, 32);
#pragma unroll
        for (int r = 0; r < 8; ++r) l_run[r] = l_run[r] * alpha[r] + ls[r];

#pragma unroll
        for (int j = 0; j < 4; ++j)
#pragma unroll
            for (int r = 0; r < 8; ++r) oacc[j][r] *= alpha[r];

        // ---- O += P V ----
        Frag16 ap[2];
#pragma unroll
        for (int kt = 0; kt < 2; ++kt) {
            const unsigned short* row = pw + lmod * 64 + kt * 32;
            ushortx8 c0 = *(const ushortx8*)(row + lhalf * 8);
            ushortx8 c1 = *(const ushortx8*)(row + 16 + lhalf * 8);
#pragma unroll
            for (int e = 0; e < 8; ++e) { ap[kt].u[e] = c0[e]; ap[kt].u[8 + e] = c1[e]; }
        }
#pragma unroll
        for (int kt = 0; kt < 2; ++kt) {
#pragma unroll
            for (int j = 0; j < 4; ++j) {
                Frag16 bv;  // B[k=lk, n=h] = V^T[h][lk]
                const unsigned short* row =
                    vbp + (j * 16 + lmod) * 64 + kt * 32;
                ushortx8 c0 = *(const ushortx8*)(row + lhalf * 8);
                ushortx8 c1 = *(const ushortx8*)(row + 16 + lhalf * 8);
#pragma unroll
                for (int e = 0; e < 8; ++e) { bv.u[e] = c0[e]; bv.u[8 + e] = c1[e]; }
                oacc[j] = wmma_bf16(ap[kt], bv, oacc[j]);
            }
        }
        __syncthreads();    // all waves done with buffer cur before re-stage
    }

    // ---- normalize and store bf16 O into [B, L, D] with col = head*64 + h ----
    float inv[8];
#pragma unroll
    for (int r = 0; r < 8; ++r) inv[r] = 1.0f / l_run[r];
#pragma unroll
    for (int j = 0; j < 4; ++j) {
        const int n = head * 64 + j * 16 + lmod;
#pragma unroll
        for (int r = 0; r < 8; ++r) {
            const int lq = qBase + lhalf * 8 + r;
            O[((size_t)b * 2048 + lq) * 1024 + n] = f2bf(oacc[j][r] * inv[r]);
        }
    }
}

// ---------------------------------------------------------------------------
// Host launcher. Workspace layout (bf16 elements):
//   xb[4M] wqb[1M] wkb[256K] wvb[256K] wob[1M] | Q[4M] K[1M] V^T[1M] O[4M]
// ---------------------------------------------------------------------------
extern "C" void kernel_launch(void* const* d_in, const int* in_sizes, int n_in,
                              void* d_out, int out_size, void* d_ws, size_t ws_size,
                              hipStream_t stream) {
    (void)in_sizes; (void)n_in; (void)out_size; (void)ws_size;
    const float* x    = (const float*)d_in[0];
    const int*   mask = (const int*)d_in[1];
    const float* Wq   = (const float*)d_in[2];
    const float* Wk   = (const float*)d_in[3];
    const float* Wv   = (const float*)d_in[4];
    const float* Wo   = (const float*)d_in[5];
    float* out = (float*)d_out;

    unsigned short* ws = (unsigned short*)d_ws;
    const size_t XE  = (size_t)2 * 2048 * 1024;     // 4,194,304
    const size_t WQE = (size_t)1024 * 1024;         // 1,048,576
    const size_t WKE = (size_t)256 * 1024;          //   262,144
    const size_t QE  = XE;
    const size_t KE  = (size_t)2 * 4 * 2048 * 64;   // 1,048,576

    unsigned short* xb  = ws;
    unsigned short* wqb = xb  + XE;
    unsigned short* wkb = wqb + WQE;
    unsigned short* wvb = wkb + WKE;
    unsigned short* wob = wvb + WKE;
    unsigned short* qws = wob + WQE;
    unsigned short* kws = qws + QE;
    unsigned short* vws = kws + KE;
    unsigned short* ows = vws + KE;

    // 0) one-time f32 -> bf16 conversions.
    gqa_cvt_kernel<<<(XE / 4 + 255) / 256, 256, 0, stream>>>(x, xb, (int)(XE / 4));
    gqa_cvt_kernel<<<(WQE / 4 + 255) / 256, 256, 0, stream>>>(Wq, wqb, (int)(WQE / 4));
    gqa_cvt_kernel<<<(WKE / 4 + 255) / 256, 256, 0, stream>>>(Wk, wkb, (int)(WKE / 4));
    gqa_cvt_kernel<<<(WKE / 4 + 255) / 256, 256, 0, stream>>>(Wv, wvb, (int)(WKE / 4));
    gqa_cvt_kernel<<<(WQE / 4 + 255) / 256, 256, 0, stream>>>(Wo, wob, (int)(WQE / 4));

    // 1) projections (M = B*L = 4096).
    gqa_gemm_kernel<<<dim3(8, 32), 256, 0, stream>>>(xb, wqb, qws, 1024, 16, 0);
    gqa_gemm_kernel<<<dim3(2, 32), 256, 0, stream>>>(xb, wkb, kws, 256, 4, 0);
    gqa_gemm_kernel<<<dim3(2, 32), 256, 0, stream>>>(xb, wvb, vws, 256, 4, 1);

    // 2) attention: (query chunk of 128, head, batch).
    gqa_attn_kernel<<<dim3(16, 16, 2), 256, 0, stream>>>(qws, kws, vws, mask, ows);

    // 3) output projection -> f32 result.
    gqa_gemm_kernel<<<dim3(8, 32), 256, 0, stream>>>(ows, wob, out, 1024, 16, 2);
}
